// MetalPKA_GNN_88914412961900
// MI455X (gfx1250) — matrix operand order repaired and checked by reading, not compiled, not run on hardware
//
#include <hip/hip_runtime.h>
#include <hip/hip_fp16.h>

typedef __attribute__((ext_vector_type(16))) _Float16 v16h;
typedef __attribute__((ext_vector_type(8)))  _Float16 v8h;
typedef __attribute__((ext_vector_type(8)))  float    v8f;

#define N_NODES 16384
#define N_EDGES 262144
#define NODE_DIM 78
#define BOND_DIM 11
#define HID 64
#define HEADS 4
#define QKV (HEADS * HID)   // 256
#define DEPTH 4
#define GEMM_WAVES 4

__device__ __forceinline__ v16h cat8(v8h lo, v8h hi) {
  return __builtin_shufflevector(lo, hi, 0, 1, 2, 3, 4, 5, 6, 7,
                                 8, 9, 10, 11, 12, 13, 14, 15);
}

// ---------------- helpers: ordered-uint float max encoding ----------------
__device__ __forceinline__ unsigned enc_max(float f) {
  unsigned b = __float_as_uint(f);
  return (b & 0x80000000u) ? ~b : (b | 0x80000000u);
}
__device__ __forceinline__ float dec_max(unsigned k) {
  unsigned b = (k & 0x80000000u) ? (k & 0x7FFFFFFFu) : ~k;
  return __uint_as_float(b);
}

// ---------------- pack f32 weight [K=64, N] into WMMA B-fragment order ----------------
// Packed layout: Bp[(((nT*(K/32) + c)*32 + lane)*16 + i]
//   lane 0-15 : col = nT*16 + lane,     K = c*32 + i        (i = 0..15)
//   lane 16-31: col = nT*16 + lane-16,  K = c*32 + 16 + i
__global__ void k_pack_b(const float* __restrict__ B, _Float16* __restrict__ Bp, int N) {
  int idx = blockIdx.x * blockDim.x + threadIdx.x;     // over N * HID
  if (idx >= N * HID) return;
  int i    = idx & 15;
  int lane = (idx >> 4) & 31;
  int c    = (idx >> 9) & 1;                           // K/32 == 2 chunks
  int nT   = idx >> 10;                                // 1024 packed elems per 16-col tile
  int col  = nT * 16 + (lane & 15);
  int k    = c * 32 + (lane >> 4) * 16 + i;
  Bp[idx] = (_Float16)B[(size_t)k * N + col];
}

// ---------------- node projection: h = x @ Wp + bp  (K=78, scalar) ----------------
__global__ void k_node_proj(const float* __restrict__ x, const float* __restrict__ Wp,
                            const float* __restrict__ bp, float* __restrict__ h,
                            _Float16* __restrict__ h16) {
  __shared__ float xr[NODE_DIM];
  int n = blockIdx.x;
  int c = threadIdx.x;                                 // 0..63
  for (int j = c; j < NODE_DIM; j += HID) xr[j] = x[n * NODE_DIM + j];
  __syncthreads();
  float acc = bp[c];
  #pragma unroll 6
  for (int j = 0; j < NODE_DIM; ++j) acc += xr[j] * Wp[j * HID + c];
  h[n * HID + c] = acc;
  h16[n * HID + c] = (_Float16)acc;
}

// ---------------- WMMA f16 GEMM: C[16384, N] = A[16384, 64] @ B[64, N] + bias ----------
// FLAGS: 1 = write f32 C, 2 = write f16 C, 4 = relu. B is pre-packed (k_pack_b).
template <int FLAGS>
__global__ void k_wmma_gemm(const _Float16* __restrict__ A, const _Float16* __restrict__ Bp,
                            const float* __restrict__ bias, float* __restrict__ C32,
                            _Float16* __restrict__ C16, int N) {
  constexpr int K = HID;                               // 64
  int lane = threadIdx.x & 31;
  int wave = threadIdx.x >> 5;
  int mT = blockIdx.x;
  int nT = blockIdx.y * GEMM_WAVES + wave;
  if (nT * 16 >= N) return;                            // wave-uniform: EXEC all-ones for WMMA
  int l15    = lane & 15;
  int hiHalf = lane >> 4;
  int rowA = mT * 16 + l15;
  int colC = nT * 16 + l15;
  int kbA  = hiHalf * 16;                              // A: K runs {0..15}/{16..31}

  v8f acc = {};
  #pragma unroll
  for (int c = 0; c < K / 32; ++c) {
    const _Float16* ap = A + (size_t)rowA * K + c * 32 + kbA;
    v16h a = cat8(*(const v8h*)ap, *(const v8h*)(ap + 8));
    const _Float16* bptr = Bp + (((size_t)nT * (K / 32) + c) * 32 + lane) * 16;
    v16h b = cat8(*(const v8h*)bptr, *(const v8h*)(bptr + 8));
    acc = __builtin_amdgcn_wmma_f32_16x16x32_f16(false, a, false, b, (short)0, acc,
                                                 false, false);
  }
  float bv = bias[colC];
  #pragma unroll
  for (int i = 0; i < 8; ++i) {
    int r = mT * 16 + i + hiHalf * 8;                  // C: VGPR i -> row i + 8*(lane>=16)
    float v = acc[i] + bv;
    if (FLAGS & 4) v = v > 0.0f ? v : 0.0f;
    size_t o = (size_t)r * N + colC;
    if (FLAGS & 1) C32[o] = v;
    if (FLAGS & 2) C16[o] = (_Float16)v;
  }
}

// ---------------- per-depth attention state init ----------------
__global__ void k_attn_init(float* __restrict__ agg, float* __restrict__ denom,
                            unsigned* __restrict__ amax) {
  int i = blockIdx.x * blockDim.x + threadIdx.x;
  if (i < N_NODES * QKV) agg[i] = 0.0f;
  if (i < N_NODES * HEADS) { denom[i] = 0.0f; amax[i] = 0u; }
}

// ---------------- edge pass A: alpha + segment max ----------------
// e-projection recomputed on the fly from LDS-staged We/be
// (avoids a 268MB [E,256] f32 intermediate that cannot stay L2-resident).
// Grid is exact: N_EDGES*HEADS % 256 == 0.
__global__ void k_edge_alpha(const _Float16* __restrict__ q16, const _Float16* __restrict__ k16,
                             const float* __restrict__ ea, const float* __restrict__ We,
                             const float* __restrict__ be,
                             const int* __restrict__ src, const int* __restrict__ dst,
                             float* __restrict__ alpha, unsigned* __restrict__ amax) {
  __shared__ float sW[BOND_DIM * QKV + QKV];
  for (int t = threadIdx.x; t < BOND_DIM * QKV + QKV; t += blockDim.x)
    sW[t] = (t < BOND_DIM * QKV) ? We[t] : be[t - BOND_DIM * QKV];
  __syncthreads();
  int idx = blockIdx.x * blockDim.x + threadIdx.x;
  int e = idx >> 2, h = idx & 3;
  int s = src[e], d = dst[e];
  float eb[BOND_DIM];
  #pragma unroll
  for (int b = 0; b < BOND_DIM; ++b) eb[b] = ea[e * BOND_DIM + b];
  const v8h* qp = (const v8h*)(q16 + (size_t)d * QKV + h * HID);
  const v8h* kp = (const v8h*)(k16 + (size_t)s * QKV + h * HID);
  const float* Wc = sW + h * HID;
  const float* bc = sW + BOND_DIM * QKV + h * HID;
  float a = 0.0f;
  #pragma unroll 2
  for (int c8 = 0; c8 < HID / 8; ++c8) {
    v8h qv = qp[c8], kv = kp[c8];
    #pragma unroll
    for (int j = 0; j < 8; ++j) {
      int dd = c8 * 8 + j;
      float ev = bc[dd];
      #pragma unroll
      for (int b = 0; b < BOND_DIM; ++b) ev += eb[b] * Wc[b * QKV + dd];
      a += (float)qv[j] * ((float)kv[j] + ev);
    }
  }
  a *= 0.125f;                                         // 1/sqrt(HID)
  alpha[idx] = a;
  atomicMax(&amax[d * HEADS + h], enc_max(a));
}

// ---------------- edge pass B: exp + segment sum ----------------
__global__ void k_edge_exp(const int* __restrict__ dst, const unsigned* __restrict__ amax,
                           float* __restrict__ alpha, float* __restrict__ denom) {
  int idx = blockIdx.x * blockDim.x + threadIdx.x;
  int e = idx >> 2, h = idx & 3;
  int d = dst[e];
  float ex = __expf(alpha[idx] - dec_max(amax[d * HEADS + h]));
  alpha[idx] = ex;                                     // reuse buffer for ex
  atomicAdd(&denom[d * HEADS + h], ex);
}

// ---------------- edge pass C: weighted aggregate scatter ----------------
__global__ void k_edge_agg(const _Float16* __restrict__ v16p, const float* __restrict__ ea,
                           const float* __restrict__ We, const float* __restrict__ be,
                           const int* __restrict__ src, const int* __restrict__ dst,
                           const float* __restrict__ ex, const float* __restrict__ denom,
                           float* __restrict__ agg) {
  __shared__ float sW[BOND_DIM * QKV + QKV];
  for (int t = threadIdx.x; t < BOND_DIM * QKV + QKV; t += blockDim.x)
    sW[t] = (t < BOND_DIM * QKV) ? We[t] : be[t - BOND_DIM * QKV];
  __syncthreads();
  int idx = blockIdx.x * blockDim.x + threadIdx.x;
  int e = idx >> 2, h = idx & 3;
  int s = src[e], d = dst[e];
  float attn = ex[idx] / (denom[d * HEADS + h] + 1e-16f);
  float eb[BOND_DIM];
  #pragma unroll
  for (int b = 0; b < BOND_DIM; ++b) eb[b] = ea[e * BOND_DIM + b];
  const v8h* vp = (const v8h*)(v16p + (size_t)s * QKV + h * HID);
  const float* Wc = sW + h * HID;
  const float* bc = sW + BOND_DIM * QKV + h * HID;
  float* ap = agg + (size_t)d * QKV + h * HID;
  #pragma unroll 2
  for (int c8 = 0; c8 < HID / 8; ++c8) {
    v8h vv = vp[c8];
    #pragma unroll
    for (int j = 0; j < 8; ++j) {
      int dd = c8 * 8 + j;
      float ev = bc[dd];
      #pragma unroll
      for (int b = 0; b < BOND_DIM; ++b) ev += eb[b] * Wc[b * QKV + dd];
      atomicAdd(&ap[dd], ((float)vv[j] + ev) * attn);
    }
  }
}

// ---------------- m = mean_heads(agg) + skip ----------------
__global__ void k_combine(const float* __restrict__ agg, const float* __restrict__ skip,
                          _Float16* __restrict__ m16) {
  int i = blockIdx.x * blockDim.x + threadIdx.x;
  if (i >= N_NODES * HID) return;
  int n = i >> 6, dd = i & 63;
  const float* a = agg + (size_t)n * QKV;
  float m = 0.25f * (a[dd] + a[HID + dd] + a[2 * HID + dd] + a[3 * HID + dd]) + skip[i];
  m16[i] = (_Float16)m;
}

// ---------------- h = h + tanh(g) ----------------
__global__ void k_hupdate(const float* __restrict__ g, float* __restrict__ h,
                          _Float16* __restrict__ h16) {
  int i = blockIdx.x * blockDim.x + threadIdx.x;
  if (i >= N_NODES * HID) return;
  float v = h[i] + tanhf(g[i]);
  h[i] = v;
  h16[i] = (_Float16)v;
}

// ---------------- final heads: logits (128 -> 2) and pka (128 -> 1) ----------------
__global__ void k_heads(const _Float16* __restrict__ t1, const _Float16* __restrict__ t3,
                        const float* __restrict__ W2, const float* __restrict__ b2,
                        const float* __restrict__ W4, const float* __restrict__ b4,
                        float* __restrict__ out) {
  int n = blockIdx.x * blockDim.x + threadIdx.x;
  if (n >= N_NODES) return;
  const v8h* r1 = (const v8h*)(t1 + (size_t)n * 128);
  const v8h* r3 = (const v8h*)(t3 + (size_t)n * 128);
  float l0 = b2[0], l1 = b2[1], p = b4[0];
  #pragma unroll 4
  for (int c8 = 0; c8 < 16; ++c8) {
    v8h a1 = r1[c8], a3 = r3[c8];
    #pragma unroll
    for (int j = 0; j < 8; ++j) {
      int jj = c8 * 8 + j;
      l0 += (float)a1[j] * W2[jj * 2 + 0];
      l1 += (float)a1[j] * W2[jj * 2 + 1];
      p  += (float)a3[j] * W4[jj];
    }
  }
  out[n * 2 + 0] = l0;
  out[n * 2 + 1] = l1;
  out[2 * N_NODES + n] = p * 3.0f + 7.0f;
}

extern "C" void kernel_launch(void* const* d_in, const int* in_sizes, int n_in,
                              void* d_out, int out_size, void* d_ws, size_t ws_size,
                              hipStream_t stream) {
  const float* x  = (const float*)d_in[0];
  const float* ea = (const float*)d_in[1];
  const int*   ei = (const int*)d_in[2];
  const float* Wp = (const float*)d_in[3];  const float* bp = (const float*)d_in[4];
  const float* Wq = (const float*)d_in[5];  const float* bq = (const float*)d_in[6];
  const float* Wk = (const float*)d_in[7];  const float* bk = (const float*)d_in[8];
  const float* Wv = (const float*)d_in[9];  const float* bv = (const float*)d_in[10];
  const float* We = (const float*)d_in[11]; const float* be = (const float*)d_in[12];
  const float* Ws = (const float*)d_in[13]; const float* bs = (const float*)d_in[14];
  const float* Wg = (const float*)d_in[15]; const float* bg = (const float*)d_in[16];
  const float* W1 = (const float*)d_in[17]; const float* b1 = (const float*)d_in[18];
  const float* W2 = (const float*)d_in[19]; const float* b2 = (const float*)d_in[20];
  const float* W3 = (const float*)d_in[21]; const float* b3 = (const float*)d_in[22];
  const float* W4 = (const float*)d_in[23]; const float* b4 = (const float*)d_in[24];
  const int* srcI = ei;
  const int* dstI = ei + N_EDGES;
  float* out = (float*)d_out;

  // -------- workspace carve-out --------
  char* w = (char*)d_ws;
  size_t off = 0;
  auto alloc = [&](size_t bytes) -> void* {
    void* p = w + off;
    off = (off + bytes + 255) & ~(size_t)255;
    return p;
  };
  float*    h     = (float*)   alloc((size_t)N_NODES * HID * 4);
  _Float16* h16   = (_Float16*)alloc((size_t)N_NODES * HID * 2);
  _Float16* q16   = (_Float16*)alloc((size_t)N_NODES * QKV * 2);
  _Float16* kk16  = (_Float16*)alloc((size_t)N_NODES * QKV * 2);
  _Float16* vv16  = (_Float16*)alloc((size_t)N_NODES * QKV * 2);
  float*    s32   = (float*)   alloc((size_t)N_NODES * HID * 4);
  _Float16* m16   = (_Float16*)alloc((size_t)N_NODES * HID * 2);
  float*    g32   = (float*)   alloc((size_t)N_NODES * HID * 4);
  float*    alpha = (float*)   alloc((size_t)N_EDGES * HEADS * 4);
  float*    denom = (float*)   alloc((size_t)N_NODES * HEADS * 4);
  unsigned* amax  = (unsigned*)alloc((size_t)N_NODES * HEADS * 4);
  float*    agg   = (float*)   alloc((size_t)N_NODES * QKV * 4);
  _Float16* t1    = (_Float16*)alloc((size_t)N_NODES * 128 * 2);
  _Float16* t3    = (_Float16*)alloc((size_t)N_NODES * 128 * 2);
  _Float16* Wq16  = (_Float16*)alloc(HID * QKV * 2);   // packed fragment order
  _Float16* Wk16  = (_Float16*)alloc(HID * QKV * 2);
  _Float16* Wv16  = (_Float16*)alloc(HID * QKV * 2);
  _Float16* Ws16  = (_Float16*)alloc(HID * HID * 2);
  _Float16* Wg16  = (_Float16*)alloc(HID * HID * 2);
  _Float16* W116  = (_Float16*)alloc(HID * 128 * 2);
  _Float16* W316  = (_Float16*)alloc(HID * 128 * 2);

  // -------- pack GEMM weights into WMMA fragment order --------
  k_pack_b<<<(HID * QKV + 255) / 256, 256, 0, stream>>>(Wq, Wq16, QKV);
  k_pack_b<<<(HID * QKV + 255) / 256, 256, 0, stream>>>(Wk, Wk16, QKV);
  k_pack_b<<<(HID * QKV + 255) / 256, 256, 0, stream>>>(Wv, Wv16, QKV);
  k_pack_b<<<(HID * HID + 255) / 256, 256, 0, stream>>>(Ws, Ws16, HID);
  k_pack_b<<<(HID * HID + 255) / 256, 256, 0, stream>>>(Wg, Wg16, HID);
  k_pack_b<<<(HID * 128 + 255) / 256, 256, 0, stream>>>(W1, W116, 128);
  k_pack_b<<<(HID * 128 + 255) / 256, 256, 0, stream>>>(W3, W316, 128);

  // -------- node projection --------
  k_node_proj<<<N_NODES, HID, 0, stream>>>(x, Wp, bp, h, h16);

  const int MT = N_NODES / 16;                         // 1024 M-tiles
  dim3 blkG(32 * GEMM_WAVES);                          // 4 waves / block
  const int EHB = (N_EDGES * HEADS) / 256;             // exact
  const int NHB = (N_NODES * HID + 255) / 256;

  for (int depth = 0; depth < DEPTH; ++depth) {
    // q/k/v projections -> f16 [N, 256]
    k_wmma_gemm<2><<<dim3(MT, QKV / (16 * GEMM_WAVES)), blkG, 0, stream>>>(
        h16, Wq16, bq, nullptr, q16, QKV);
    k_wmma_gemm<2><<<dim3(MT, QKV / (16 * GEMM_WAVES)), blkG, 0, stream>>>(
        h16, Wk16, bk, nullptr, kk16, QKV);
    k_wmma_gemm<2><<<dim3(MT, QKV / (16 * GEMM_WAVES)), blkG, 0, stream>>>(
        h16, Wv16, bv, nullptr, vv16, QKV);

    // segment-softmax attention over edges
    k_attn_init<<<(N_NODES * QKV + 255) / 256, 256, 0, stream>>>(agg, denom, amax);
    k_edge_alpha<<<EHB, 256, 0, stream>>>(q16, kk16, ea, We, be, srcI, dstI, alpha, amax);
    k_edge_exp<<<EHB, 256, 0, stream>>>(dstI, amax, alpha, denom);
    k_edge_agg<<<EHB, 256, 0, stream>>>(vv16, ea, We, be, srcI, dstI, alpha, denom, agg);

    // skip GEMM, combine, gate GEMM, residual tanh update
    k_wmma_gemm<1><<<dim3(MT, 1), blkG, 0, stream>>>(h16, Ws16, bs, s32, nullptr, HID);
    k_combine<<<NHB, 256, 0, stream>>>(agg, s32, m16);
    k_wmma_gemm<1><<<dim3(MT, 1), blkG, 0, stream>>>(m16, Wg16, bg, g32, nullptr, HID);
    k_hupdate<<<NHB, 256, 0, stream>>>(g32, h, h16);
  }

  // -------- output heads --------
  k_wmma_gemm<6><<<dim3(MT, 128 / (16 * GEMM_WAVES)), blkG, 0, stream>>>(
      h16, W116, b1, nullptr, t1, 128);
  k_wmma_gemm<6><<<dim3(MT, 128 / (16 * GEMM_WAVES)), blkG, 0, stream>>>(
      h16, W316, b3, nullptr, t3, 128);
  k_heads<<<(N_NODES + 255) / 256, 256, 0, stream>>>(t1, t3, W2, b2, W4, b4, out);

  (void)in_sizes; (void)n_in; (void)out_size; (void)ws_size;
}
